// PyGGraphLayer_16054587752806
// MI455X (gfx1250) — matrix operand order repaired and checked by reading, not compiled, not run on hardware
//
#include <hip/hip_runtime.h>

typedef __attribute__((ext_vector_type(16))) _Float16 v16h;
typedef __attribute__((ext_vector_type(8)))  _Float16 v8h;
typedef __attribute__((ext_vector_type(8)))  float    v8f;

#define J_   24
#define D_   256
#define H_   4
#define DH_  64
#define EMAX 96

// ---------------------------------------------------------------------------
// Prep: Wt[n][k] = (f16) W[k][n]   (W is [D, H*Dh] row-major, 256x256)
// 128 KB result lives in d_ws; stays resident in the 192MB L2 for all blocks.
// ---------------------------------------------------------------------------
__global__ void gat_prep_wt(const float* __restrict__ W, _Float16* __restrict__ Wt) {
    int idx = blockIdx.x * 256 + threadIdx.x;   // 0..65535
    int k = idx >> 8;
    int n = idx & 255;
    Wt[n * 256 + k] = (_Float16)W[idx];
}

// ---------------------------------------------------------------------------
// Fused GAT layer: one workgroup per (b,t) graph. 256 threads = 8 wave32.
// Phase 1: x[b,t] (24x256 f32) -> f16 LDS (padded to 32 rows)
// Phase 2: h = x @ W via v_wmma_f32_16x16x32_f16 (f32 accum), h -> LDS
// Phase 3: GAT attention (chain + self loops) + aggregation + bias
// ---------------------------------------------------------------------------
__global__ void __launch_bounds__(256)
gat_fused(const float* __restrict__ x,
          const _Float16* __restrict__ Wt,
          const float* __restrict__ att_src,
          const float* __restrict__ att_dst,
          const float* __restrict__ bias,
          const int*   __restrict__ ei,     // [2, E0] row-major
          float*       __restrict__ out,
          int E0)
{
    __shared__ _Float16 xlds[32 * 256];          // 16 KB  (A operand, f16)
    __shared__ float    hlds[J_ * 256];          // 24 KB  (h, f32)
    __shared__ float    as_l[J_ * H_], ad_l[J_ * H_];
    __shared__ float    m_l [J_ * H_], dn_l[J_ * H_];
    __shared__ float    esc [EMAX * H_];
    __shared__ int      esrc[EMAX], edst[EMAX];
    __shared__ int      degl[J_], adj[J_ * 8];

    const int tid = threadIdx.x;
    const int g   = blockIdx.x;                  // graph id = b*T + t
    const int E   = E0 + J_;                     // edges + self loops

    // ---- Phase 1: stage x -> f16 LDS ------------------------------------
    {
        const float4* xg = (const float4*)(x + (size_t)g * (J_ * D_));
        #pragma unroll
        for (int c = 0; c < 6; ++c) {            // 6144 floats / (256 thr * 4)
            float4 v = xg[c * 256 + tid];
            int f = c * 1024 + tid * 4;
            xlds[f + 0] = (_Float16)v.x;
            xlds[f + 1] = (_Float16)v.y;
            xlds[f + 2] = (_Float16)v.z;
            xlds[f + 3] = (_Float16)v.w;
        }
        v8h z = {};
        *(v8h*)&xlds[24 * 256 + tid * 8] = z;    // zero rows 24..31
    }
    __syncthreads();

    // ---- Phase 2: GEMM via WMMA -----------------------------------------
    // 2 M-tiles x 16 N-tiles; wave w owns m = w&1, n-tiles (w>>1)*4 .. +3.
    const int wave = tid >> 5, lane = tid & 31;
    const int mt = wave & 1;
    const int nb = (wave >> 1) * 4;
    const int lm = lane & 15, lh = lane >> 4;

    v8f zero = {};
    v8f acc0 = zero, acc1 = zero, acc2 = zero, acc3 = zero;

    const int arow = (mt * 16 + lm) * 256;
    for (int kk = 0; kk < 256; kk += 32) {
        // A fragment: 16x32 f16 (ISA layout: lane=M, two 8-elem K chunks)
        union { v16h v; v8h h[2]; } af;
        af.h[0] = *(const v8h*)&xlds[arow + kk + 8 * lh];
        af.h[1] = *(const v8h*)&xlds[arow + kk + 8 * lh + 16];

        #pragma unroll
        for (int j = 0; j < 4; ++j) {
            // B fragment: 32x16 f16, lane=N, two 8-elem K chunks (same as A).
            // Wt is N-major so each lane reads two 16-byte K chunks (L2 hit).
            const _Float16* bp = Wt + ((nb + j) * 16 + lm) * 256 + kk + 8 * lh;
            union { v16h v; v8h h[2]; } bf;
            bf.h[0] = *(const v8h*)bp;
            bf.h[1] = *(const v8h*)(bp + 16);

            v8f c = (j == 0) ? acc0 : (j == 1) ? acc1 : (j == 2) ? acc2 : acc3;
            c = __builtin_amdgcn_wmma_f32_16x16x32_f16(
                    false, af.v, false, bf.v, (short)0, c, false, false);
            if (j == 0) acc0 = c; else if (j == 1) acc1 = c;
            else if (j == 2) acc2 = c; else acc3 = c;
        }
    }

    // D layout: VGPR r -> row = mt*16 + 8*(lane>>4) + r, col = ntile*16 + lane&15
    #pragma unroll
    for (int j = 0; j < 4; ++j) {
        v8f c = (j == 0) ? acc0 : (j == 1) ? acc1 : (j == 2) ? acc2 : acc3;
        #pragma unroll
        for (int r = 0; r < 8; ++r) {
            int row = mt * 16 + lh * 8 + r;
            if (row < J_)
                hlds[row * 256 + (nb + j) * 16 + lm] = c[r];
        }
    }
    __syncthreads();

    // ---- Phase 3a: a_src/a_dst dot products + edge list -----------------
    if (tid < J_ * H_) {
        int j = tid >> 2, h = tid & 3;
        const float* hp = &hlds[j * 256 + h * DH_];
        const float* ps = att_src + h * DH_;
        const float* pd = att_dst + h * DH_;
        float ss = 0.f, sd = 0.f;
        #pragma unroll 8
        for (int d = 0; d < DH_; ++d) {
            float hv = hp[d];
            ss += hv * ps[d];
            sd += hv * pd[d];
        }
        as_l[tid] = ss;
        ad_l[tid] = sd;
    }
    for (int e = tid; e < E; e += 256) {
        int s, d;
        if (e < E0) { s = ei[e]; d = ei[E0 + e]; }
        else        { s = e - E0; d = e - E0; }   // self loops
        esrc[e] = s;
        edst[e] = d;
    }
    __syncthreads();

    // ---- Phase 3b: incoming-edge CSR per node ---------------------------
    if (tid < J_) {
        int dg = 0;
        for (int e = 0; e < E; ++e)
            if (edst[e] == tid && dg < 8) adj[tid * 8 + dg++] = e;
        degl[tid] = dg;
    }
    __syncthreads();

    // ---- Phase 3c: leaky-relu logits ------------------------------------
    for (int p = tid; p < E * H_; p += 256) {
        int e = p >> 2, h = p & 3;
        float s = as_l[esrc[e] * H_ + h] + ad_l[edst[e] * H_ + h];
        esc[p] = (s >= 0.f) ? s : 0.2f * s;
    }
    __syncthreads();

    // ---- Phase 3d: per-(dst,head) max & denom (no atomics needed) -------
    if (tid < J_ * H_) {
        int j = tid >> 2, h = tid & 3;
        int dg = degl[j];
        float mx = -3.402823466e38f;
        for (int s = 0; s < dg; ++s)
            mx = fmaxf(mx, esc[adj[j * 8 + s] * H_ + h]);
        float dn = 0.f;
        for (int s = 0; s < dg; ++s)
            dn += __expf(esc[adj[j * 8 + s] * H_ + h] - mx);
        m_l[tid] = mx;
        dn_l[tid] = dn;
    }
    __syncthreads();

    // ---- Phase 3e: alpha ------------------------------------------------
    for (int p = tid; p < E * H_; p += 256) {
        int e = p >> 2, h = p & 3;
        int d = edst[e];
        esc[p] = __expf(esc[p] - m_l[d * H_ + h]) / dn_l[d * H_ + h];
    }
    __syncthreads();

    // ---- Phase 3f: aggregate + bias, coalesced store --------------------
    {
        int h = tid >> 6;                         // head of this column
        float bv = bias[tid];
        float* op = out + (size_t)g * (J_ * D_);
        for (int j = 0; j < J_; ++j) {
            float a = bv;
            int dg = degl[j];
            for (int s = 0; s < dg; ++s) {
                int e = adj[j * 8 + s];
                a += esc[e * H_ + h] * hlds[esrc[e] * 256 + tid];
            }
            op[j * 256 + tid] = a;
        }
    }
}

// ---------------------------------------------------------------------------
extern "C" void kernel_launch(void* const* d_in, const int* in_sizes, int n_in,
                              void* d_out, int out_size, void* d_ws, size_t ws_size,
                              hipStream_t stream) {
    const float* x       = (const float*)d_in[0];
    const float* W       = (const float*)d_in[1];
    const float* att_src = (const float*)d_in[2];
    const float* att_dst = (const float*)d_in[3];
    const float* bias    = (const float*)d_in[4];
    const int*   ei      = (const int*)d_in[5];

    int E0 = in_sizes[5] / 2;                     // 46
    int nGraphs = in_sizes[0] / (J_ * D_);        // B*T = 4096

    _Float16* Wt = (_Float16*)d_ws;               // 128 KB scratch

    gat_prep_wt<<<256, 256, 0, stream>>>(W, Wt);
    gat_fused<<<nGraphs, 256, 0, stream>>>(x, Wt, att_src, att_dst, bias, ei,
                                           (float*)d_out, E0);
}
